// MoE_predictor_54339926229148
// MI455X (gfx1250) — matrix operand order repaired and checked by reading, not compiled, not run on hardware
//
#include <hip/hip_runtime.h>
#include <hip/hip_bf16.h>

// ---------------------------------------------------------------- constants
#define T_TOK 16384   // B*S
#define HD    256     // hidden
#define NE    8       // experts
#define TDIM  768     // text dim
#define IDIM  1024    // image dim

typedef _Float16 f16;
typedef __attribute__((ext_vector_type(4)))  float    v4f;
typedef __attribute__((ext_vector_type(8)))  _Float16 v8h;
typedef __attribute__((ext_vector_type(16))) _Float16 v16h;
typedef __attribute__((ext_vector_type(8)))  float    v8f;
typedef __attribute__((ext_vector_type(4)))  unsigned int u32x4;
typedef __attribute__((ext_vector_type(8)))  unsigned int u32x8;

__device__ __forceinline__ float gelu_f(float x) {
  // exact erf GELU (torch nn.GELU default)
  return 0.5f * x * (1.0f + erff(x * 0.70710678118654752440f));
}

// ---- WMMA fragment loaders (LDS, 16B-aligned b128 loads) -------------------
// A 16x32 f16: lane m(0..15) row m; kb=(lane>=16)?8:0; a[j]=A[m][kb+j],
// a[8+j]=A[m][16+kb+j]   (CDNA5 ISA 7.12.2)
__device__ __forceinline__ v16h ld_afrag(const f16* base, int stride) {
  int lane = threadIdx.x & 31;
  int m    = lane & 15;
  int kb   = (lane >> 4) << 3;
  const f16* p = base + m * stride + kb;
  v8h lo = *(const v8h*)(p);
  v8h hi = *(const v8h*)(p + 16);
  return __builtin_shufflevector(lo, hi, 0,1,2,3,4,5,6,7,8,9,10,11,12,13,14,15);
}
// B 32x16 f16 from LDS stored transposed Bt[n][k]: lane n col n;
// kb=(lane>=16)?8:0; b[j]=B[kb+j][n], b[8+j]=B[16+kb+j][n]
__device__ __forceinline__ v16h ld_bfrag(const f16* bt, int stride) {
  int lane = threadIdx.x & 31;
  int n    = lane & 15;
  int kb   = (lane >> 4) << 3;
  const f16* p = bt + n * stride + kb;
  v8h lo = *(const v8h*)(p);
  v8h hi = *(const v8h*)(p + 16);
  return __builtin_shufflevector(lo, hi, 0,1,2,3,4,5,6,7,8,9,10,11,12,13,14,15);
}

__device__ __forceinline__ v8f wmma16(v16h a, v16h b, v8f c) {
  return __builtin_amdgcn_wmma_f32_16x16x32_f16(false, a, false, b, (short)0, c,
                                                false, false);
}

// ---- Tensor Data Mover: DMA a 2D tile (rows_n x 32 f16) of a row-major
// [rows][row_len_k] f16 matrix into LDS with +16B padding per row (stride 40).
// D# per CDNA5 ISA; 2-group form (tile is 2-D).
__device__ __forceinline__ void tdm_issue_bt(const f16* gsrc, const f16* lds_dst,
                                             int row_len_k, int rows_n) {
  unsigned long long ga = (unsigned long long)gsrc;
  u32x4 g0;
  g0[0] = 1u;                                        // count=1, user descriptor
  g0[1] = (unsigned)(unsigned long long)lds_dst;     // lds_addr (low 32 bits)
  g0[2] = (unsigned)ga;                              // global_addr[31:0]
  g0[3] = (unsigned)((ga >> 32) & 0x01ffffffULL)     // global_addr[56:32]
          | (2u << 30);                              // type=2 ("image")
  u32x8 g1;
  g1[0] = (1u << 16)            // data_size = 1 -> 2 bytes
        | (1u << 20)            // pad_enable
        | (3u << 22)            // pad_interval = 3 -> every 16 DWORDs (64B row)
        | (3u << 25);           // pad_amount = 3 -> 4 DWORDs = 16B (stride 40 f16)
  g1[1] = ((unsigned)row_len_k & 0xffffu) << 16;                 // tensor_dim0 lo
  g1[2] = (((unsigned)row_len_k >> 16) & 0xffffu)                // tensor_dim0 hi
        | (((unsigned)rows_n & 0xffffu) << 16);                  // tensor_dim1 lo
  g1[3] = (((unsigned)rows_n >> 16) & 0xffffu)                   // tensor_dim1 hi
        | (32u << 16);                                           // tile_dim0 = 32
  g1[4] = (unsigned)rows_n & 0xffffu;                            // tile_dim1
  g1[5] = (unsigned)row_len_k;                                   // dim0_stride lo
  g1[6] = 0u;                                                    // stride hi
  g1[7] = 0u;
  asm volatile("tensor_load_to_lds %0, %1" :: "s"(g0), "s"(g1) : "memory");
}
__device__ __forceinline__ void tdm_wait() {
  __builtin_amdgcn_s_wait_tensorcnt(0);
}

// ---- A-tile staging: 16 fp32 -> 16 f16 into LDS (2x ds_store_b128) ---------
__device__ __forceinline__ void stage_a16(const float* src, f16* dst) {
  const v4f* s4 = (const v4f*)src;
  v4f a0 = s4[0], a1 = s4[1], a2 = s4[2], a3 = s4[3];
  v8h h0, h1;
#pragma unroll
  for (int j = 0; j < 4; ++j) {
    h0[j] = (f16)a0[j]; h0[4 + j] = (f16)a1[j];
    h1[j] = (f16)a2[j]; h1[4 + j] = (f16)a3[j];
  }
  *(v8h*)dst = h0;
  *(v8h*)(dst + 8) = h1;
}

// ----------------------------------------- fp32 [K][N] -> f16 [N][K] (batch)
__global__ __launch_bounds__(256)
void cvt_transpose_f16(const float* __restrict__ s, f16* __restrict__ d,
                       int K, int N, int total) {
  int i = blockIdx.x * 256 + threadIdx.x;
  if (i >= total) return;
  int per = K * N;
  int e = i / per, r = i - e * per;
  int n = r / K, k = r - n * K;
  d[i] = (f16)s[(size_t)e * per + (size_t)k * N + n];
}

// ------------------------------------------------ embed: gelu(x@We+be)+emb
// tile 128(M) x 64(N), K=768; weights pre-transposed Wt[N=256][K=768]
// double-buffered LDS: TDM(i+1) + A-stage(i+1) overlap WMMA(i)
__global__ __launch_bounds__(256)
void embed_gemm_kernel(const float* __restrict__ A, const f16* __restrict__ Wt,
                       const float* __restrict__ bias,
                       const float* __restrict__ l2e, const float* __restrict__ cle,
                       float* __restrict__ O1, float* __restrict__ O2) {
  __shared__ f16 As[2][128 * 40];
  __shared__ f16 Bt[2][64 * 40];
  const int m0 = blockIdx.x * 128, n0 = blockIdx.y * 64;
  const int tid = threadIdx.x, lane = tid & 31, wid = tid >> 5;
  const int row = tid >> 1, half = tid & 1;
  const float* arow = A + (size_t)(m0 + row) * TDIM + half * 16;
  v8f z = 0.0f;
  v8f acc[4] = {z, z, z, z};
  const int NK = TDIM / 32;
  // prologue: fill buffer 0
  if (wid == 0) tdm_issue_bt(Wt + (size_t)n0 * TDIM, Bt[0], TDIM, 64);
  stage_a16(arow, As[0] + row * 40 + half * 16);
  for (int i = 0; i < NK; ++i) {
    const int cur = i & 1, nxt = cur ^ 1;
    if (wid == 0) tdm_wait();
    __syncthreads();
    if (i + 1 < NK) {  // prefetch next tile into alternate buffers
      if (wid == 0)
        tdm_issue_bt(Wt + (size_t)n0 * TDIM + (i + 1) * 32, Bt[nxt], TDIM, 64);
      stage_a16(arow + (i + 1) * 32, As[nxt] + row * 40 + half * 16);
    }
    v16h af = ld_afrag(As[cur] + wid * 16 * 40, 40);
#pragma unroll
    for (int f = 0; f < 4; ++f)
      acc[f] = wmma16(af, ld_bfrag(Bt[cur] + f * 16 * 40, 40), acc[f]);
  }
  const int colb = n0 + (lane & 15);
  const int rowb = m0 + wid * 16 + ((lane >> 4) << 3);
#pragma unroll
  for (int f = 0; f < 4; ++f) {
    int col = colb + f * 16;
    float bv = bias[col], e1 = l2e[col], e2 = cle[col];
#pragma unroll
    for (int r = 0; r < 8; ++r) {
      float g = gelu_f(acc[f][r] + bv);
      size_t o = (size_t)(rowb + r) * HD + col;
      O1[o] = g + e1;
      O2[o] = g + e2;
    }
  }
}

// ------------------------------------------------------------------ gating
__global__ __launch_bounds__(256)
void gate_kernel(const float* __restrict__ in, const float* __restrict__ gW,
                 const float* __restrict__ gb, int* __restrict__ counts,
                 int* __restrict__ rows, float* __restrict__ wlist) {
  int lane = threadIdx.x & 31, wid = threadIdx.x >> 5;
  int t = blockIdx.x * 8 + wid;
  float acc[NE];
#pragma unroll
  for (int e = 0; e < NE; ++e) acc[e] = 0.0f;
  const float* xr = in + (size_t)t * HD;
  for (int h = lane; h < HD; h += 32) {
    float xv = xr[h];
#pragma unroll
    for (int e = 0; e < NE; ++e) acc[e] += xv * gW[h * NE + e];
  }
#pragma unroll
  for (int off = 16; off > 0; off >>= 1)
#pragma unroll
    for (int e = 0; e < NE; ++e) acc[e] += __shfl_xor(acc[e], off, 32);
  if (lane == 0) {
    float lg[NE], mx = -1e30f;
#pragma unroll
    for (int e = 0; e < NE; ++e) { lg[e] = acc[e] + gb[e]; mx = fmaxf(mx, lg[e]); }
    float s = 0.0f;
#pragma unroll
    for (int e = 0; e < NE; ++e) { lg[e] = expf(lg[e] - mx); s += lg[e]; }
    float inv = 1.0f / s;
    int i0 = 0; float v0 = -1.0f;
#pragma unroll
    for (int e = 0; e < NE; ++e) if (lg[e] > v0) { v0 = lg[e]; i0 = e; }
    int i1 = 0; float v1 = -1.0f;
#pragma unroll
    for (int e = 0; e < NE; ++e) if (e != i0 && lg[e] > v1) { v1 = lg[e]; i1 = e; }
    int p0 = atomicAdd(&counts[i0], 1);
    rows[i0 * T_TOK + p0] = t; wlist[i0 * T_TOK + p0] = v0 * inv;
    int p1 = atomicAdd(&counts[i1], 1);
    rows[i1 * T_TOK + p1] = t; wlist[i1 * T_TOK + p1] = v1 * inv;
  }
}

// --------------------------------------------- fused expert FFN (gathered)
// block = (tile of 128 gathered rows) x expert; N = 256 full width
// weights pre-transposed Wt[e][N][K]; B tiles via TDM, double-buffered
__global__ __launch_bounds__(256)
void expert_kernel(const float* __restrict__ in, const f16* __restrict__ W1t,
                   const f16* __restrict__ W2t, const float* __restrict__ b1,
                   const float* __restrict__ b2, const int* __restrict__ counts,
                   const int* __restrict__ rows, const float* __restrict__ wlist,
                   float* __restrict__ moe_out) {
  extern __shared__ char smem[];
  f16* hbuf = (f16*)smem;                  // 128 x 264
  f16* As0  = hbuf + 128 * 264;            // 2 x (128 x 40)
  f16* As1  = As0 + 128 * 40;
  f16* Bt0  = As1 + 128 * 40;              // 2 x (256 x 40)
  f16* Bt1  = Bt0 + 256 * 40;
  f16* Asb[2] = {As0, As1};
  f16* Btb[2] = {Bt0, Bt1};
  const int e = blockIdx.y;
  const int cnt = counts[e];
  const int m0 = blockIdx.x * 128;
  if (m0 >= cnt) return;
  const int tid = threadIdx.x, lane = tid & 31, wid = tid >> 5;
  const int* rlist = rows + e * T_TOK;
  const int row = tid >> 1, half = tid & 1;
  const int gr = m0 + row;
  const float* arow =
      (gr < cnt) ? (in + (size_t)rlist[gr] * HD + half * 16) : (const float*)0;
  v8f z = 0.0f;
  v8f acc[16];
#pragma unroll
  for (int f = 0; f < 16; ++f) acc[f] = z;

  auto stageA = [&](int i, f16* buf) {
    f16* dst = buf + row * 40 + half * 16;
    if (arow) {
      stage_a16(arow + i * 32, dst);
    } else {
      v8h zh = (f16)0.0f;
      *(v8h*)dst = zh; *(v8h*)(dst + 8) = zh;
    }
  };

  // ---------- GEMM1: gathered A @ W1[e]   (8 k-steps, double-buffered)
  const f16* W1e = W1t + (size_t)e * HD * HD;
  if (wid == 0) tdm_issue_bt(W1e, Bt0, HD, HD);
  stageA(0, As0);
  for (int i = 0; i < HD / 32; ++i) {
    const int cur = i & 1, nxt = cur ^ 1;
    if (wid == 0) tdm_wait();
    __syncthreads();
    if (i + 1 < HD / 32) {
      if (wid == 0) tdm_issue_bt(W1e + (i + 1) * 32, Btb[nxt], HD, HD);
      stageA(i + 1, Asb[nxt]);
    }
    v16h af = ld_afrag(Asb[cur] + wid * 16 * 40, 40);
#pragma unroll
    for (int f = 0; f < 16; ++f)
      acc[f] = wmma16(af, ld_bfrag(Btb[cur] + f * 16 * 40, 40), acc[f]);
  }
  // epilogue1: +b1, gelu, -> LDS h (f16)
  const int nlane = lane & 15;
  const int rbase = wid * 16 + ((lane >> 4) << 3);
#pragma unroll
  for (int f = 0; f < 16; ++f) {
    int col = f * 16 + nlane;
    float bv = b1[e * HD + col];
#pragma unroll
    for (int r = 0; r < 8; ++r)
      hbuf[(rbase + r) * 264 + col] = (f16)gelu_f(acc[f][r] + bv);
  }
  __syncthreads();

  // ---------- GEMM2: h @ W2[e]   (double-buffered TDM only; A is in LDS)
#pragma unroll
  for (int f = 0; f < 16; ++f) acc[f] = z;
  const f16* W2e = W2t + (size_t)e * HD * HD;
  if (wid == 0) tdm_issue_bt(W2e, Bt0, HD, HD);
  for (int i = 0; i < HD / 32; ++i) {
    const int cur = i & 1, nxt = cur ^ 1;
    if (wid == 0) tdm_wait();
    __syncthreads();
    if (i + 1 < HD / 32 && wid == 0)
      tdm_issue_bt(W2e + (i + 1) * 32, Btb[nxt], HD, HD);
    v16h af = ld_afrag(hbuf + wid * 16 * 264 + i * 32, 264);
#pragma unroll
    for (int f = 0; f < 16; ++f)
      acc[f] = wmma16(af, ld_bfrag(Btb[cur] + f * 16 * 40, 40), acc[f]);
  }
  // epilogue2: (+b2) * gate_weight, atomic scatter (exactly 2 adds/elem total)
#pragma unroll
  for (int f = 0; f < 16; ++f) {
    int col = f * 16 + nlane;
    float bv = b2[e * HD + col];
#pragma unroll
    for (int r = 0; r < 8; ++r) {
      int g = m0 + rbase + r;
      if (g < cnt) {
        int t = rlist[g];
        float w = wlist[e * T_TOK + g];
        atomicAdd(&moe_out[(size_t)t * HD + col], (acc[f][r] + bv) * w);
      }
    }
  }
}

// ---------------------------------- layernorm + gelu + residual (in place)
__global__ __launch_bounds__(256)
void ln_gelu_res_kernel(float* __restrict__ y, const float* __restrict__ resid,
                        const float* __restrict__ g, const float* __restrict__ bb) {
  int lane = threadIdx.x & 31, wid = threadIdx.x >> 5;
  int t = blockIdx.x * 8 + wid;
  float v[8], s = 0.0f;
#pragma unroll
  for (int j = 0; j < 8; ++j) { v[j] = y[(size_t)t * HD + j * 32 + lane]; s += v[j]; }
#pragma unroll
  for (int off = 16; off > 0; off >>= 1) s += __shfl_xor(s, off, 32);
  float m = s * (1.0f / 256.0f);
  float q = 0.0f;
#pragma unroll
  for (int j = 0; j < 8; ++j) { float d = v[j] - m; q += d * d; }
#pragma unroll
  for (int off = 16; off > 0; off >>= 1) q += __shfl_xor(q, off, 32);
  float inv = rsqrtf(q * (1.0f / 256.0f) + 1e-5f);
#pragma unroll
  for (int j = 0; j < 8; ++j) {
    int h = j * 32 + lane;
    float zz = (v[j] - m) * inv * g[h] + bb[h];
    y[(size_t)t * HD + h] = gelu_f(zz) + resid[(size_t)t * HD + h];
  }
}

// ----------------------------------------------- head GEMM: A@W + bias
// weights pre-transposed Wt[N][K=256]; double-buffered
__global__ __launch_bounds__(256)
void head_gemm_kernel(const float* __restrict__ A, const f16* __restrict__ Wt,
                      const float* __restrict__ bias, float* __restrict__ C,
                      int N) {
  __shared__ f16 As[2][128 * 40];
  __shared__ f16 Bt[2][64 * 40];
  const int m0 = blockIdx.x * 128, n0 = blockIdx.y * 64;
  const int tid = threadIdx.x, lane = tid & 31, wid = tid >> 5;
  const int row = tid >> 1, half = tid & 1;
  const float* arow = A + (size_t)(m0 + row) * HD + half * 16;
  v8f z = 0.0f;
  v8f acc[4] = {z, z, z, z};
  if (wid == 0) tdm_issue_bt(Wt + (size_t)n0 * HD, Bt[0], HD, 64);
  stage_a16(arow, As[0] + row * 40 + half * 16);
  for (int i = 0; i < HD / 32; ++i) {
    const int cur = i & 1, nxt = cur ^ 1;
    if (wid == 0) tdm_wait();
    __syncthreads();
    if (i + 1 < HD / 32) {
      if (wid == 0)
        tdm_issue_bt(Wt + (size_t)n0 * HD + (i + 1) * 32, Bt[nxt], HD, 64);
      stage_a16(arow + (i + 1) * 32, As[nxt] + row * 40 + half * 16);
    }
    v16h af = ld_afrag(As[cur] + wid * 16 * 40, 40);
#pragma unroll
    for (int f = 0; f < 4; ++f)
      acc[f] = wmma16(af, ld_bfrag(Bt[cur] + f * 16 * 40, 40), acc[f]);
  }
  const int colb = n0 + (lane & 15);
  const int rowb = m0 + wid * 16 + ((lane >> 4) << 3);
#pragma unroll
  for (int f = 0; f < 4; ++f) {
    int col = colb + f * 16;
    float bv = bias[col];
#pragma unroll
    for (int r = 0; r < 8; ++r)
      C[(size_t)(rowb + r) * N + col] = acc[f][r] + bv;
  }
}

// ===========================================================================
extern "C" void kernel_launch(void* const* d_in, const int* in_sizes, int n_in,
                              void* d_out, int out_size, void* d_ws, size_t ws_size,
                              hipStream_t stream) {
  const float* x      = (const float*)d_in[0];
  const float* We     = (const float*)d_in[1];
  const float* be     = (const float*)d_in[2];
  const float* l2_emb = (const float*)d_in[3];
  const float* cl_emb = (const float*)d_in[4];
  const float* gate_W = (const float*)d_in[5];
  const float* gate_b = (const float*)d_in[6];
  const float* W1     = (const float*)d_in[7];
  const float* b1     = (const float*)d_in[8];
  const float* W2     = (const float*)d_in[9];
  const float* b2     = (const float*)d_in[10];
  const float* l2_g   = (const float*)d_in[11];
  const float* l2_b   = (const float*)d_in[12];
  const float* cl_g   = (const float*)d_in[13];
  const float* cl_b   = (const float*)d_in[14];
  const float* Wt2i   = (const float*)d_in[15];
  const float* bt2i   = (const float*)d_in[16];
  const float* Wcl    = (const float*)d_in[17];
  const float* bcl    = (const float*)d_in[18];
  float* out = (float*)d_out;

  // ---- workspace layout (256B aligned) ----
  char* ws = (char*)d_ws;
  size_t off = 0;
  auto take = [&](size_t bytes) -> char* {
    char* p = ws + off;
    off = (off + bytes + 255) & ~(size_t)255;
    return p;
  };
  f16* wWe   = (f16*)take((size_t)TDIM * HD * 2);   // [256][768]
  f16* wW1   = (f16*)take((size_t)NE * HD * HD * 2);// [E][256][256] (T)
  f16* wW2   = (f16*)take((size_t)NE * HD * HD * 2);
  f16* wWt2i = (f16*)take((size_t)HD * IDIM * 2);   // [1024][256]
  f16* wWcl  = (f16*)take((size_t)HD * HD * 2);     // [256][256]
  float* l2_in = (float*)take((size_t)T_TOK * HD * 4);
  float* cl_in = (float*)take((size_t)T_TOK * HD * 4);
  int*   counts = (int*)take(2 * NE * 4);
  int*   rowsb  = (int*)take((size_t)2 * NE * T_TOK * 4);
  float* wlistb = (float*)take((size_t)2 * NE * T_TOK * 4);
  float* moe0   = (float*)take((size_t)T_TOK * HD * 4);
  float* moe1   = (float*)take((size_t)T_TOK * HD * 4);
  int*   cnt0 = counts;             int* cnt1 = counts + NE;
  int*   rows0 = rowsb;             int* rows1 = rowsb + (size_t)NE * T_TOK;
  float* wl0 = wlistb;              float* wl1 = wlistb + (size_t)NE * T_TOK;

  // ---- zero routing counts + moe accumulators (required every call) ----
  hipMemsetAsync(counts, 0, 2 * NE * sizeof(int), stream);
  hipMemsetAsync(moe0, 0, (size_t)T_TOK * HD * sizeof(float), stream);
  hipMemsetAsync(moe1, 0, (size_t)T_TOK * HD * sizeof(float), stream);

  // ---- weights fp32 [K][N] -> f16 [N][K] (transposed for TDM tiles) ----
  auto cvtT = [&](const float* s, f16* d, int K, int N, int batch) {
    int total = batch * K * N;
    cvt_transpose_f16<<<(total + 255) / 256, 256, 0, stream>>>(s, d, K, N, total);
  };
  cvtT(We,   wWe,   TDIM, HD, 1);
  cvtT(W1,   wW1,   HD, HD, NE);
  cvtT(W2,   wW2,   HD, HD, NE);
  cvtT(Wt2i, wWt2i, HD, IDIM, 1);
  cvtT(Wcl,  wWcl,  HD, HD, 1);

  // ---- embed: gelu(x@We+be) + {l2_emb, cl_emb} ----
  embed_gemm_kernel<<<dim3(T_TOK / 128, HD / 64), 256, 0, stream>>>(
      x, wWe, be, l2_emb, cl_emb, l2_in, cl_in);

  // ---- gating (both branches) ----
  gate_kernel<<<T_TOK / 8, 256, 0, stream>>>(l2_in, gate_W, gate_b, cnt0, rows0, wl0);
  gate_kernel<<<T_TOK / 8, 256, 0, stream>>>(cl_in, gate_W, gate_b, cnt1, rows1, wl1);

  // ---- fused expert FFN (gathered top-2 only) ----
  size_t shmem = (size_t)(128 * 264 + 2 * 128 * 40 + 2 * 256 * 40) * sizeof(f16);
  expert_kernel<<<dim3(T_TOK / 128, NE), 256, shmem, stream>>>(
      l2_in, wW1, wW2, b1, b2, cnt0, rows0, wl0, moe0);
  expert_kernel<<<dim3(T_TOK / 128, NE), 256, shmem, stream>>>(
      cl_in, wW1, wW2, b1, b2, cnt1, rows1, wl1, moe1);

  // ---- layernorm + gelu + residual (in place on moe buffers) ----
  ln_gelu_res_kernel<<<T_TOK / 8, 256, 0, stream>>>(moe0, l2_in, l2_g, l2_b);
  ln_gelu_res_kernel<<<T_TOK / 8, 256, 0, stream>>>(moe1, cl_in, cl_g, cl_b);

  // ---- heads -> d_out (l2_res [T,1024] then cl_res [T,256]) ----
  head_gemm_kernel<<<dim3(T_TOK / 128, IDIM / 64), 256, 0, stream>>>(
      moe0, wWt2i, bt2i, out, IDIM);
  head_gemm_kernel<<<dim3(T_TOK / 128, HD / 64), 256, 0, stream>>>(
      moe1, wWcl, bcl, out + (size_t)T_TOK * IDIM, HD);
}
